// GraphormerBlock_84679575208545
// MI455X (gfx1250) — matrix operand order repaired and checked by reading, not compiled, not run on hardware
//
#include <hip/hip_runtime.h>
#include <math.h>

// ---------------------------------------------------------------------------
// Problem constants (from reference)
// ---------------------------------------------------------------------------
#define BS   4
#define NN   128
#define DD   768
#define HH   8
#define HDD  96
#define FFN  3072
#define EE   (BS * NN * NN)      // 65536
#define NODES (BS * NN)          // 512

typedef __attribute__((ext_vector_type(16))) _Float16 v16h;
typedef __attribute__((ext_vector_type(8)))  float    v8f;

// ---------------------------------------------------------------------------
// WMMA: D = A(16x32 f16) x B(32x16 f16) + C(16x16 f32), wave32.
// Operand stripings: contiguous K (k = 16*(lane>>4) + i), same order in A and B.
// ---------------------------------------------------------------------------
__device__ __forceinline__ v8f wmma32(v16h a, v16h b, v8f c) {
  return __builtin_amdgcn_wmma_f32_16x16x32_f16(false, a, false, b,
                                                (short)0, c, false, false);
}

// A-matrix 16x32 (MxK) from f16 row-major (global or LDS).
// lane L holds row M = m0 + (L&15); lanes 0-15: K in {0..15},
// lanes 16-31: K in {16..31} (relative to k0).
__device__ __forceinline__ v16h load_A_h16(const _Float16* __restrict__ A, int lda,
                                           int m0, int k0, int lane) {
  const _Float16* p = A + (size_t)(m0 + (lane & 15)) * lda + k0 + ((lane >> 4) << 4);
  v16h a;
#pragma unroll
  for (int i = 0; i < 16; ++i) a[i] = p[i];
  return a;
}

// A-matrix from fp32 source (used only for LDS probs in attention).
__device__ __forceinline__ v16h load_A_f32(const float* __restrict__ A, int lda,
                                           int m0, int k0, int lane) {
  const float* p = A + (size_t)(m0 + (lane & 15)) * lda + k0 + ((lane >> 4) << 4);
  v16h a;
#pragma unroll
  for (int i = 0; i < 16; ++i) a[i] = (_Float16)p[i];
  return a;
}

// B-matrix 32x16 (KxN) where B(k,n) = W[n*ldw + k]  (f16 weight [N,K] row-major,
// computing X @ W^T). lane holds col N = n0+(lane&15); lanes 0-15: K=k0+0..15,
// lanes 16-31: K=k0+16..31; element i -> +i. 32B contiguous per lane.
__device__ __forceinline__ v16h load_B_wT_h16(const _Float16* __restrict__ W, int ldw,
                                              int n0, int k0, int lane) {
  const _Float16* p = W + (size_t)(n0 + (lane & 15)) * ldw + k0 + ((lane >> 4) << 4);
  v16h b;
#pragma unroll
  for (int i = 0; i < 16; ++i) b[i] = p[i];
  return b;
}

// B-matrix 32x16 (KxN) where B(k,n) = B[k*ldb + n]  (f16 [K,N] row-major).
__device__ __forceinline__ v16h load_B_rm_h16(const _Float16* __restrict__ B, int ldb,
                                              int k0, int n0, int lane) {
  const _Float16* p = B + (size_t)(k0 + ((lane >> 4) << 4)) * ldb + n0 + (lane & 15);
  v16h b;
#pragma unroll
  for (int i = 0; i < 16; ++i) b[i] = p[(size_t)i * ldb];
  return b;
}

__device__ __forceinline__ float silu_f(float x) {
  return x / (1.0f + __expf(-x));
}

// ---------------------------------------------------------------------------
// fp32 -> f16 conversion (weights, once per launch; L2-resident afterwards)
// ---------------------------------------------------------------------------
__global__ void cvt_f16_kernel(const float* __restrict__ in,
                               _Float16* __restrict__ out, int n) {
  int i = blockIdx.x * blockDim.x + threadIdx.x;
  if (i < n) out[i] = (_Float16)in[i];
}

// De-stride cm1 [768,1538] into packed f16 row-part and col-part [768,768].
__global__ void cvt_cm1_kernel(const float* __restrict__ cm1_w,
                               _Float16* __restrict__ row16,
                               _Float16* __restrict__ col16) {
  int idx = blockIdx.x * blockDim.x + threadIdx.x;
  if (idx >= DD * DD) return;
  int r = idx / DD, c = idx - r * DD;
  row16[idx] = (_Float16)cm1_w[(size_t)r * 1538 + c];
  col16[idx] = (_Float16)cm1_w[(size_t)r * 1538 + DD + c];
}

// ---------------------------------------------------------------------------
// LayerNorm over rows of 768 -> f16 output (feeds WMMA A operands)
// ---------------------------------------------------------------------------
__global__ void ln_kernel(const float* __restrict__ in,
                          const float* __restrict__ s,
                          const float* __restrict__ b,
                          _Float16* __restrict__ out, int cols) {
  int row = blockIdx.x;
  const float* p = in + (size_t)row * cols;
  float sum = 0.f, sq = 0.f;
  for (int c = threadIdx.x; c < cols; c += blockDim.x) {
    float v = p[c]; sum += v; sq += v * v;
  }
  __shared__ float rs[8], rq[8];
#pragma unroll
  for (int off = 16; off; off >>= 1) {
    sum += __shfl_down(sum, off, 32);
    sq  += __shfl_down(sq,  off, 32);
  }
  int wid = threadIdx.x >> 5, lid = threadIdx.x & 31;
  if (lid == 0) { rs[wid] = sum; rq[wid] = sq; }
  __syncthreads();
  if (wid == 0) {
    sum = (lid < 8) ? rs[lid] : 0.f;
    sq  = (lid < 8) ? rq[lid] : 0.f;
#pragma unroll
    for (int off = 4; off; off >>= 1) {
      sum += __shfl_down(sum, off, 32);
      sq  += __shfl_down(sq,  off, 32);
    }
    if (lid == 0) { rs[0] = sum; rq[0] = sq; }
  }
  __syncthreads();
  float mean = rs[0] / cols;
  float var  = rq[0] / cols - mean * mean;
  float inv  = rsqrtf(var + 1e-5f);
  for (int c = threadIdx.x; c < cols; c += blockDim.x)
    out[(size_t)row * cols + c] = (_Float16)((p[c] - mean) * inv * s[c] + b[c]);
}

// ---------------------------------------------------------------------------
// f16 GEMM, register-blocked: one wave computes a 16(M) x 64(N) tile.
// Per K-step: 1 A load feeds 4 independent WMMA chains (4 accumulators).
// grid = (N/64, M/16), block = 32 (one wave, EXEC all ones for WMMA).
// ---------------------------------------------------------------------------
__global__ void gemm_h16_kernel(const _Float16* __restrict__ A, int lda,
                                const _Float16* __restrict__ W, int ldw,
                                const float* __restrict__ bias,
                                const float* __restrict__ res,
                                float* __restrict__ C32,
                                _Float16* __restrict__ C16,
                                int ldc, int K) {
  int lane = threadIdx.x;
  int n0 = blockIdx.x << 6;      // 64-wide N tile
  int m0 = blockIdx.y << 4;
  v8f acc0 = {}, acc1 = {}, acc2 = {}, acc3 = {};
  for (int k0 = 0; k0 < K; k0 += 32) {
    v16h a  = load_A_h16(A, lda, m0, k0, lane);
    v16h b0 = load_B_wT_h16(W, ldw, n0,      k0, lane);
    v16h b1 = load_B_wT_h16(W, ldw, n0 + 16, k0, lane);
    v16h b2 = load_B_wT_h16(W, ldw, n0 + 32, k0, lane);
    v16h b3 = load_B_wT_h16(W, ldw, n0 + 48, k0, lane);
    acc0 = wmma32(a, b0, acc0);
    acc1 = wmma32(a, b1, acc1);
    acc2 = wmma32(a, b2, acc2);
    acc3 = wmma32(a, b3, acc3);
  }
  int nl = lane & 15;
  int mbase = m0 + ((lane >> 4) << 3);
  v8f accs[4] = {acc0, acc1, acc2, acc3};
#pragma unroll
  for (int u = 0; u < 4; ++u) {
    int n = n0 + u * 16 + nl;
    float bn = bias ? bias[n] : 0.f;
#pragma unroll
    for (int r = 0; r < 8; ++r) {
      int m = mbase + r;
      float v = accs[u][r] + bn;
      if (res) v += res[(size_t)m * ldc + n];
      if (C32) C32[(size_t)m * ldc + n] = v;
      if (C16) C16[(size_t)m * ldc + n] = (_Float16)v;
    }
  }
}

// ---------------------------------------------------------------------------
// Edge geometry + per-head edge bias / gate (tiny 2->8 matmuls, pure VALU)
// ---------------------------------------------------------------------------
__global__ void edge_geom_kernel(const float* __restrict__ x,
                                 const float* __restrict__ edge_attr,
                                 const float* __restrict__ edge_mask,
                                 const float* __restrict__ efc1_w,
                                 const float* __restrict__ efc1_b,
                                 const float* __restrict__ efc2_w,
                                 const float* __restrict__ efc2_b,
                                 float* __restrict__ radial_o,
                                 float* __restrict__ cdiff_o,
                                 float* __restrict__ attn_bias,
                                 float* __restrict__ tanhgw) {
  int e = blockIdx.x * blockDim.x + threadIdx.x;
  if (e >= EE) return;
  int b = e >> 14, loc = e & 16383, i = loc >> 7, j = loc & 127;
  int rn = b * NN + i, cn = b * NN + j;
  float dx = x[rn * 3 + 0] - x[cn * 3 + 0];
  float dy = x[rn * 3 + 1] - x[cn * 3 + 1];
  float dz = x[rn * 3 + 2] - x[cn * 3 + 2];
  float rad = dx * dx + dy * dy + dz * dz;
  float inv = 1.0f / (sqrtf(rad + 1e-8f) + 1.0f);
  radial_o[e] = rad;
  cdiff_o[e * 3 + 0] = dx * inv;
  cdiff_o[e * 3 + 1] = dy * inv;
  cdiff_o[e * 3 + 2] = dz * inv;
  float em = edge_mask[e], ea1 = edge_attr[e];
#pragma unroll
  for (int h = 0; h < HH; ++h) {
    float y1 = (rad * efc1_w[h * 2] + ea1 * efc1_w[h * 2 + 1] + efc1_b[h]) * em;
    float y2 = (rad * efc2_w[h * 2] + ea1 * efc2_w[h * 2 + 1] + efc2_b[h]) * em;
    size_t o = (((size_t)b * HH + h) * NN + i) * NN + j;
    attn_bias[o] = y1;
    tanhgw[o] = tanhf(y2);
  }
}

// ---------------------------------------------------------------------------
// Attention: one block per (b,h). Scores (128x128 fp32) live in 64KB LDS.
// scores = (q@k^T)*hd^-.5 + bias ; probs = tanh(gw)*softmax ; att = probs@v
// Register-blocked: q's A operand reused across all 8 score col-blocks;
// probs' A operand reused across all 6 output HD-blocks.
// ---------------------------------------------------------------------------
__global__ void attention_kernel(const _Float16* __restrict__ qkv16,
                                 const float* __restrict__ attn_bias,
                                 const float* __restrict__ tanhgw,
                                 _Float16* __restrict__ att16) {
  __shared__ float sc[NN * NN];  // 64 KB
  int bh = blockIdx.x;
  int b = bh >> 3, h = bh & 7;
  const _Float16* qb = qkv16 + (size_t)(b * NN) * (3 * DD) + h * HDD;
  const _Float16* kb = qb + DD;
  const _Float16* vb = qb + 2 * DD;
  const float* biasb = attn_bias + (size_t)bh * NN * NN;
  const float* gateb = tanhgw + (size_t)bh * NN * NN;
  int wave = threadIdx.x >> 5, lane = threadIdx.x & 31;
  const float scale = 0.1020620726159658f;  // 96^-0.5, applied post-WMMA

  // scores: wave w owns row-block w; 8 independent acc chains, A loaded once
  {
    v8f acc[8] = {};
    for (int k0 = 0; k0 < HDD; k0 += 32) {
      v16h a = load_A_h16(qb, 3 * DD, wave * 16, k0, lane);
#pragma unroll
      for (int cb = 0; cb < 8; ++cb) {
        v16h bm = load_B_wT_h16(kb, 3 * DD, cb * 16, k0, lane);
        acc[cb] = wmma32(a, bm, acc[cb]);
      }
    }
    int nl = lane & 15;
    int mbase = wave * 16 + ((lane >> 4) << 3);
#pragma unroll
    for (int cb = 0; cb < 8; ++cb) {
      int n = cb * 16 + nl;
#pragma unroll
      for (int r = 0; r < 8; ++r) {
        int m = mbase + r;
        sc[m * NN + n] = acc[cb][r] * scale + biasb[m * NN + n];
      }
    }
  }
  __syncthreads();

  // gated softmax: one thread per row
  if (threadIdx.x < NN) {
    float* rowp = sc + threadIdx.x * NN;
    const float* gp = gateb + (size_t)threadIdx.x * NN;
    float mx = -1e30f;
    for (int j = 0; j < NN; ++j) mx = fmaxf(mx, rowp[j]);
    float sum = 0.f;
    for (int j = 0; j < NN; ++j) { float ev = __expf(rowp[j] - mx); rowp[j] = ev; sum += ev; }
    float invs = 1.0f / sum;
    for (int j = 0; j < NN; ++j) rowp[j] = rowp[j] * invs * gp[j];
  }
  __syncthreads();

  // att = probs @ v : wave w owns row-block w; 6 HD col-blocks, A loaded once
  {
    v8f acc[6] = {};
    for (int k0 = 0; k0 < NN; k0 += 32) {
      v16h a = load_A_f32(sc, NN, wave * 16, k0, lane);
#pragma unroll
      for (int cb = 0; cb < 6; ++cb) {
        v16h bm = load_B_rm_h16(vb, 3 * DD, k0, cb * 16, lane);
        acc[cb] = wmma32(a, bm, acc[cb]);
      }
    }
    int nl = lane & 15;
    int mbase = wave * 16 + ((lane >> 4) << 3);
#pragma unroll
    for (int cb = 0; cb < 6; ++cb) {
      int n = cb * 16 + nl;
#pragma unroll
      for (int r = 0; r < 8; ++r) {
        int m = mbase + r;
        att16[(size_t)(b * NN + m) * DD + h * HDD + n] = (_Float16)acc[cb][r];
      }
    }
  }
}

// ---------------------------------------------------------------------------
// GLU (a * relu(g)^2) + exact GELU  (after fc1); f16 in -> f16 out
// ---------------------------------------------------------------------------
__global__ void glu_gelu_kernel(const _Float16* __restrict__ ff1,
                                _Float16* __restrict__ out) {
  int idx = blockIdx.x * blockDim.x + threadIdx.x;
  if (idx >= NODES * FFN) return;
  int m = idx / FFN, n = idx - m * FFN;
  float a = (float)ff1[(size_t)m * (2 * FFN) + n];
  float g = (float)ff1[(size_t)m * (2 * FFN) + FFN + n];
  float rg = fmaxf(g, 0.f);
  float u = a * rg * rg;
  out[idx] = (_Float16)(0.5f * u * (1.0f + erff(u * 0.70710678118654752f)));
}

// GLU + residual + node mask -> h_out (fp32 to d_out, f16 for cm1 GEMMs)
__global__ void glu2_res_kernel(const _Float16* __restrict__ ff2,
                                const float* __restrict__ h1,
                                const float* __restrict__ node_mask,
                                float* __restrict__ hout,
                                _Float16* __restrict__ hout16) {
  int idx = blockIdx.x * blockDim.x + threadIdx.x;
  if (idx >= NODES * DD) return;
  int m = idx / DD, n = idx - m * DD;
  float a = (float)ff2[(size_t)m * (2 * DD) + n];
  float g = (float)ff2[(size_t)m * (2 * DD) + DD + n];
  float rg = fmaxf(g, 0.f);
  float v = (h1[idx] + a * rg * rg) * node_mask[m];
  hout[idx] = v;
  hout16[idx] = (_Float16)v;
}

// ---------------------------------------------------------------------------
// Fused per-edge MLP + coordinate update.
// One block per row node (b,i): its 128 edges are contiguous (j = 0..127).
//   t1 = silu(Arow[rn] + Acol[cn] + radial*Wea0 + eattr*Wea1)   (LDS, f16)
//   t2 = silu(t1 @ cm2^T + b2) ; s = t2 . cm3   (WMMA vs f16 cm2, L2-resident)
//   agg = sum_j coord_diff * s * edge_mask ;  x_out written directly.
// Register-blocked phase 2: per K-step one LDS A load feeds 6 WMMA chains.
// ---------------------------------------------------------------------------
__global__ void edge_mlp_kernel(const float* __restrict__ Arow,
                                const float* __restrict__ Acol,
                                const float* __restrict__ radial,
                                const float* __restrict__ edge_attr,
                                const float* __restrict__ cm1_w,     // [768,1538]
                                const _Float16* __restrict__ cm2w16, // [768,768] f16
                                const float* __restrict__ cm2_b,
                                const float* __restrict__ cm3_w,     // [768]
                                const float* __restrict__ cdiff,
                                const float* __restrict__ edge_mask,
                                const float* __restrict__ x,
                                const float* __restrict__ linker_mask,
                                const float* __restrict__ node_mask,
                                float* __restrict__ xout) {
  __shared__ _Float16 t1[32 * DD];   // 48 KB: one 32-edge chunk of activations
  __shared__ float s_part[32];
  __shared__ float aggv[3];
  int rn = blockIdx.x;
  int b = rn >> 7;
  size_t ebase = (size_t)rn * NN;
  int tid = threadIdx.x, wave = tid >> 5, lane = tid & 31;
  if (tid < 3) aggv[tid] = 0.f;
  const float* arow = Arow + (size_t)rn * DD;

  for (int cc = 0; cc < 4; ++cc) {            // 4 chunks of 32 edges
    if (tid < 32) s_part[tid] = 0.f;
    __syncthreads();

    // phase 1: build t1 chunk in LDS (f16)
    for (int idx = tid; idx < 32 * DD; idx += 256) {
      int c = idx / DD, k = idx - c * DD;
      int j = cc * 32 + c;
      size_t e = ebase + j;
      int cn = b * NN + j;
      float v = arow[k] + Acol[(size_t)cn * DD + k]
              + radial[e]    * cm1_w[(size_t)k * 1538 + 1536]
              + edge_attr[e] * cm1_w[(size_t)k * 1538 + 1537];
      t1[idx] = (_Float16)silu_f(v);
    }
    __syncthreads();

    // phase 2: (32 x 768) @ cm2^T via WMMA; wave owns 6 consecutive n-blocks,
    // A loaded once per K-step from LDS, 6 independent acc chains.
    int nb0 = wave * 6;
    for (int mb = 0; mb < 2; ++mb) {
      v8f acc[6] = {};
      for (int k0 = 0; k0 < DD; k0 += 32) {
        v16h a = load_A_h16(t1, DD, mb * 16, k0, lane);
#pragma unroll
        for (int u = 0; u < 6; ++u) {
          v16h bm = load_B_wT_h16(cm2w16, DD, (nb0 + u) * 16, k0, lane);
          acc[u] = wmma32(a, bm, acc[u]);
        }
      }
      int nl = lane & 15;
      int mbase = mb * 16 + ((lane >> 4) << 3);
#pragma unroll
      for (int u = 0; u < 6; ++u) {
        int n = (nb0 + u) * 16 + nl;
        float bn = cm2_b[n], w3 = cm3_w[n];
#pragma unroll
        for (int r = 0; r < 8; ++r) {
          int c = mbase + r;
          float t2 = silu_f(acc[u][r] + bn);
          atomicAdd(&s_part[c], t2 * w3);      // ds_add_f32
        }
      }
    }
    __syncthreads();

    // phase 3: accumulate coordinate aggregation for these 32 edges
    if (tid < 32) {
      int j = cc * 32 + tid;
      size_t e = ebase + j;
      float s = s_part[tid] * edge_mask[e];
      atomicAdd(&aggv[0], cdiff[e * 3 + 0] * s);
      atomicAdd(&aggv[1], cdiff[e * 3 + 1] * s);
      atomicAdd(&aggv[2], cdiff[e * 3 + 2] * s);
    }
    __syncthreads();
  }

  if (tid < 3) {
    float a = aggv[tid] * 0.01f;               // / NORM_FACTOR
    xout[(size_t)rn * 3 + tid] =
        (x[rn * 3 + tid] + a * linker_mask[rn]) * node_mask[rn];
  }
}

// ---------------------------------------------------------------------------
// Launch
// ---------------------------------------------------------------------------
extern "C" void kernel_launch(void* const* d_in, const int* in_sizes, int n_in,
                              void* d_out, int out_size, void* d_ws, size_t ws_size,
                              hipStream_t stream) {
  (void)in_sizes; (void)n_in; (void)out_size; (void)ws_size;
  const float* h           = (const float*)d_in[0];
  const float* x           = (const float*)d_in[1];
  const float* edge_attr   = (const float*)d_in[2];
  const float* node_mask   = (const float*)d_in[3];
  const float* edge_mask   = (const float*)d_in[4];
  const float* linker_mask = (const float*)d_in[5];
  const float* in_proj_w   = (const float*)d_in[6];
  const float* in_proj_b   = (const float*)d_in[7];
  const float* out_proj_w  = (const float*)d_in[8];
  const float* out_proj_b  = (const float*)d_in[9];
  const float* ln1_s       = (const float*)d_in[10];
  const float* ln1_b       = (const float*)d_in[11];
  const float* ln2_s       = (const float*)d_in[12];
  const float* ln2_b       = (const float*)d_in[13];
  const float* fc1_w       = (const float*)d_in[14];
  const float* fc1_b       = (const float*)d_in[15];
  const float* fc2_w       = (const float*)d_in[16];
  const float* fc2_b       = (const float*)d_in[17];
  const float* efc1_w      = (const float*)d_in[18];
  const float* efc1_b      = (const float*)d_in[19];
  const float* efc2_w      = (const float*)d_in[20];
  const float* efc2_b      = (const float*)d_in[21];
  const float* cm1_w       = (const float*)d_in[22];
  const float* cm1_b       = (const float*)d_in[23];
  const float* cm2_w       = (const float*)d_in[24];
  const float* cm2_b       = (const float*)d_in[25];
  const float* cm3_w       = (const float*)d_in[26];
  // d_in[27] = edge_index: fixed fully-connected block structure, exploited.

  float* out = (float*)d_out;
  float* hout = out;                        // [512,768]
  float* xout = out + (size_t)NODES * DD;   // [512,3]

  // ---- workspace layout: fp32 region, then f16 region ----
  float* ws = (float*)d_ws;
  size_t o = 0;
  float* attn_bias = ws + o; o += (size_t)BS * HH * NN * NN; // 524288
  float* tanhgw    = ws + o; o += (size_t)BS * HH * NN * NN;
  float* h1        = ws + o; o += (size_t)NODES * DD;
  float* Arow      = ws + o; o += (size_t)NODES * DD;
  float* Acol      = ws + o; o += (size_t)NODES * DD;
  float* radial    = ws + o; o += (size_t)EE;
  float* cdiff     = ws + o; o += (size_t)EE * 3;

  _Float16* hw = (_Float16*)(ws + o);
  size_t q = 0;
  _Float16* hn16    = hw + q; q += (size_t)NODES * DD;
  _Float16* qkv16   = hw + q; q += (size_t)NODES * 3 * DD;
  _Float16* att16   = hw + q; q += (size_t)NODES * DD;
  _Float16* ln2h16  = hw + q; q += (size_t)NODES * DD;
  _Float16* ff1h    = hw + q; q += (size_t)NODES * 2 * FFN;
  _Float16* ffv16   = hw + q; q += (size_t)NODES * FFN;
  _Float16* ff2h    = hw + q; q += (size_t)NODES * 2 * DD;
  _Float16* hout16  = hw + q; q += (size_t)NODES * DD;
  _Float16* inW16   = hw + q; q += (size_t)3 * DD * DD;
  _Float16* outW16  = hw + q; q += (size_t)DD * DD;
  _Float16* fc1W16  = hw + q; q += (size_t)2 * FFN * DD;
  _Float16* fc2W16  = hw + q; q += (size_t)2 * DD * FFN;
  _Float16* cm2W16  = hw + q; q += (size_t)DD * DD;
  _Float16* cm1r16  = hw + q; q += (size_t)DD * DD;
  _Float16* cm1c16  = hw + q; q += (size_t)DD * DD;

  // ---- weight conversions (f16, L2-resident for the rest of the launch) ----
  cvt_f16_kernel<<<(3 * DD * DD + 255) / 256, 256, 0, stream>>>(in_proj_w, inW16, 3 * DD * DD);
  cvt_f16_kernel<<<(DD * DD + 255) / 256, 256, 0, stream>>>(out_proj_w, outW16, DD * DD);
  cvt_f16_kernel<<<(2 * FFN * DD + 255) / 256, 256, 0, stream>>>(fc1_w, fc1W16, 2 * FFN * DD);
  cvt_f16_kernel<<<(2 * DD * FFN + 255) / 256, 256, 0, stream>>>(fc2_w, fc2W16, 2 * DD * FFN);
  cvt_f16_kernel<<<(DD * DD + 255) / 256, 256, 0, stream>>>(cm2_w, cm2W16, DD * DD);
  cvt_cm1_kernel<<<(DD * DD + 255) / 256, 256, 0, stream>>>(cm1_w, cm1r16, cm1c16);

  // 1) LN1 -> f16
  ln_kernel<<<NODES, 256, 0, stream>>>(h, ln1_s, ln1_b, hn16, DD);
  // 2) QKV = hn @ in_proj_w^T + b  -> f16 [512,2304]
  gemm_h16_kernel<<<dim3(3 * DD / 64, NODES / 16), 32, 0, stream>>>(
      hn16, DD, inW16, DD, in_proj_b, nullptr, nullptr, qkv16, 3 * DD, DD);
  // 3) edge geometry + head bias/gate
  edge_geom_kernel<<<EE / 256, 256, 0, stream>>>(
      x, edge_attr, edge_mask, efc1_w, efc1_b, efc2_w, efc2_b,
      radial, cdiff, attn_bias, tanhgw);
  // 4) attention (one block per (b,h)) -> f16 att
  attention_kernel<<<BS * HH, 256, 0, stream>>>(qkv16, attn_bias, tanhgw, att16);
  // 5) h1 = h + att @ out_proj_w^T + b  (fp32)
  gemm_h16_kernel<<<dim3(DD / 64, NODES / 16), 32, 0, stream>>>(
      att16, DD, outW16, DD, out_proj_b, h, h1, nullptr, DD, DD);
  // 6) LN2 -> f16
  ln_kernel<<<NODES, 256, 0, stream>>>(h1, ln2_s, ln2_b, ln2h16, DD);
  // 7) ff1 = ln2h @ fc1_w^T + b -> f16 [512,6144]
  gemm_h16_kernel<<<dim3(2 * FFN / 64, NODES / 16), 32, 0, stream>>>(
      ln2h16, DD, fc1W16, DD, fc1_b, nullptr, nullptr, ff1h, 2 * FFN, DD);
  // 8) GLU + exact GELU -> f16 [512,3072]
  glu_gelu_kernel<<<(NODES * FFN + 255) / 256, 256, 0, stream>>>(ff1h, ffv16);
  // 9) ff2 = ffv @ fc2_w^T + b -> f16 [512,1536]
  gemm_h16_kernel<<<dim3(2 * DD / 64, NODES / 16), 32, 0, stream>>>(
      ffv16, FFN, fc2W16, FFN, fc2_b, nullptr, nullptr, ff2h, 2 * DD, FFN);
  // 10) GLU + residual + node_mask -> h_out (d_out) + f16 copy
  glu2_res_kernel<<<(NODES * DD + 255) / 256, 256, 0, stream>>>(
      ff2h, h1, node_mask, hout, hout16);
  // 11) cm1 decomposition: Arow = h_out @ cm1[:, :768]^T + b1 ; Acol likewise
  gemm_h16_kernel<<<dim3(DD / 64, NODES / 16), 32, 0, stream>>>(
      hout16, DD, cm1r16, DD, cm1_b, nullptr, Arow, nullptr, DD, DD);
  gemm_h16_kernel<<<dim3(DD / 64, NODES / 16), 32, 0, stream>>>(
      hout16, DD, cm1c16, DD, nullptr, nullptr, Acol, nullptr, DD, DD);
  // 12) fused per-edge MLP + coordinate update (one block per row node)
  edge_mlp_kernel<<<NODES, 256, 0, stream>>>(
      Arow, Acol, radial, edge_attr, cm1_w, cm2W16, cm2_b, cm3_w,
      cdiff, edge_mask, x, linker_mask, node_mask, xout);
}
